// SelfAttention_64665027608825
// MI455X (gfx1250) — matrix operand order repaired and checked by reading, not compiled, not run on hardware
//
#include <hip/hip_runtime.h>
#include <hip/hip_bf16.h>

// Problem constants (match reference)
constexpr int B = 8;
constexpr int C = 256;
constexpr int N = 4096;

typedef __attribute__((ext_vector_type(16))) __bf16 v16bf;
typedef __attribute__((ext_vector_type(8)))  float  v8f;

union ABu { v16bf v; unsigned int u[8]; };

__device__ __forceinline__ unsigned short f2bf(float x) {
    unsigned int u = __builtin_bit_cast(unsigned int, x);
    u = (u + 0x7FFFu + ((u >> 16) & 1u)) >> 16;   // round-to-nearest-even
    return (unsigned short)u;
}

__device__ __forceinline__ v8f wmma_bf16(const ABu& a, const ABu& b, v8f c) {
    return __builtin_amdgcn_wmma_f32_16x16x32_bf16(
        /*neg_a=*/false, a.v, /*neg_b=*/false, b.v,
        /*c_mod=*/(short)0, c, /*reuse_a=*/false, /*reuse_b=*/false);
}

// ---------------------------------------------------------------------------
// Kernel 0: pack WF (64x256), WG (64x256), WH (256x256) fp32 -> bf16 combined
// ---------------------------------------------------------------------------
__global__ void k_cvt(const float* __restrict__ WF, const float* __restrict__ WG,
                      const float* __restrict__ WH, unsigned short* __restrict__ Wbuf) {
    int i = blockIdx.x * blockDim.x + threadIdx.x;      // 0 .. 384*256-1
    if (i >= 384 * 256) return;
    int row = i >> 8, c = i & 255;
    float v;
    if (row < 64)       v = WF[row * 256 + c];
    else if (row < 128) v = WG[(row - 64) * 256 + c];
    else                v = WH[(row - 128) * 256 + c];
    Wbuf[i] = f2bf(v);
}

// ---------------------------------------------------------------------------
// Kernel 1: projections.  out[row, n] = W[row, :] . feat[:, n]  per batch.
// ---------------------------------------------------------------------------
#define FEAT_PITCH 264   // 256 + 8 shorts padding

__launch_bounds__(256)
__global__ void k_proj(const float* __restrict__ feat, const unsigned short* __restrict__ Wbuf,
                       unsigned short* __restrict__ Fbuf, unsigned short* __restrict__ Gbuf,
                       unsigned short* __restrict__ Htbuf) {
    __shared__ __align__(16) unsigned short ldsFeat[64 * FEAT_PITCH]; // [n_local][c]
    const int b     = blockIdx.x >> 6;
    const int nbase = (blockIdx.x & 63) * 64;
    const int tid   = threadIdx.x;

    for (int i = 0; i < 64; ++i) {
        int idx = tid + i * 256;
        int c = idx >> 6, j = idx & 63;
        float v = feat[(size_t)b * C * N + (size_t)c * N + nbase + j];
        ldsFeat[j * FEAT_PITCH + c] = f2bf(v);
    }
    __syncthreads();

    const int wave = tid >> 5, lane = tid & 31;
    const int l16 = lane & 15, half = lane >> 4;

    for (int t = wave; t < 96; t += 8) {
        int rt = t >> 2, ct = t & 3;
        int arow = rt * 16 + l16;
        v8f acc = {0.f, 0.f, 0.f, 0.f, 0.f, 0.f, 0.f, 0.f};
        for (int cc = 0; cc < 8; ++cc) {
            ABu a, bb;
            const unsigned short* wr = Wbuf + (size_t)arow * 256 + cc * 32;
            const unsigned short* fr = ldsFeat + (ct * 16 + l16) * FEAT_PITCH + cc * 32;
#pragma unroll
            for (int v = 0; v < 8; ++v) {
                int ka = ((v >> 2) << 4) + (half << 3) + ((v & 3) << 1);
                a.u[v] = *(const unsigned int*)(wr + ka);
                int kb = ((v >> 2) << 4) + (half << 3) + ((v & 3) << 1);
                bb.u[v] = *(const unsigned int*)(fr + kb);
            }
            acc = wmma_bf16(a, bb, acc);
        }
        int ncol = nbase + ct * 16 + l16;
        if (rt < 8) {
            unsigned short* dst = (rt < 4) ? Fbuf : Gbuf;
            int k0 = (rt & 3) * 16;
#pragma unroll
            for (int r = 0; r < 8; ++r) {
                int kd = k0 + r + (half << 3);
                dst[((size_t)b * N + ncol) * 64 + kd] = f2bf(acc[r]);
            }
        } else {
            int d0 = (rt - 8) * 16;
#pragma unroll
            for (int r = 0; r < 8; ++r) {
                int d = d0 + r + (half << 3);
                Htbuf[((size_t)b * 256 + d) * N + ncol] = f2bf(acc[r]);
            }
        }
    }
}

// ---------------------------------------------------------------------------
// Kernel 2: flash attention.  Block = (b, 128-row n tile), 128 threads = 4 waves,
// each wave owns 32 query rows (2 row-groups).  Stream m in steps of 32 with
// register-prefetch double buffering.  Row-sum computed via a constant "ones"
// B-fragment as a 17th accumulator tile.
// ---------------------------------------------------------------------------
#define F_PITCH 72   // 64 + 8 shorts
#define H_PITCH 40   // 32 + 8 shorts
#define P_PITCH 40

__launch_bounds__(128)
__global__ void k_attn(const float* __restrict__ input, const float* __restrict__ gamma_p,
                       const unsigned short* __restrict__ Fbuf,
                       const unsigned short* __restrict__ Gbuf,
                       const unsigned short* __restrict__ Htbuf,
                       float* __restrict__ out) {
    __shared__ __align__(16) unsigned short ldsF[32 * F_PITCH];     // [m_local][k]
    __shared__ __align__(16) unsigned short ldsH[256 * H_PITCH];    // [d][m_local]
    __shared__ __align__(16) unsigned short ldsP[4 * 32 * P_PITCH]; // [wave][row32][m]

    const int b     = blockIdx.x >> 5;          // N/128 = 32 row tiles
    const int nbase = (blockIdx.x & 31) * 128;
    const int tid   = threadIdx.x;
    const int wave  = tid >> 5, lane = tid & 31;
    const int l16 = lane & 15, half = lane >> 4;
    const float gamma = gamma_p[0];
    const int rowb = nbase + wave * 32;

    // G A-fragments: 2 row-groups x 2 k-chunks, resident in VGPRs for the loop
    ABu ag[2][2];
#pragma unroll
    for (int rg = 0; rg < 2; ++rg) {
        int n = rowb + rg * 16 + l16;
        const unsigned short* gp = Gbuf + ((size_t)b * N + n) * 64;
#pragma unroll
        for (int f = 0; f < 2; ++f)
#pragma unroll
            for (int v = 0; v < 8; ++v) {
                int k = f * 32 + ((v >> 2) << 4) + (half << 3) + ((v & 3) << 1);
                ag[rg][f].u[v] = *(const unsigned int*)(gp + k);
            }
    }

    v8f oacc[2][17];   // [rg][0..15] = O tiles, [rg][16] = row-sum tile
#pragma unroll
    for (int rg = 0; rg < 2; ++rg)
#pragma unroll
        for (int t = 0; t < 17; ++t) oacc[rg][t] = (v8f){0.f,0.f,0.f,0.f,0.f,0.f,0.f,0.f};
    float mrow[2][8];
#pragma unroll
    for (int rg = 0; rg < 2; ++rg)
#pragma unroll
        for (int r = 0; r < 8; ++r) mrow[rg][r] = -1e30f;

    const unsigned short* fb = Fbuf + (size_t)b * N * 64;
    const unsigned short* hb = Htbuf + (size_t)b * 256 * N;

    // Register prefetch buffers: F tile = 256 uint4 / 128 thr, H = 1024 uint4 / 128 thr
    uint4 pf[2], ph[8];
    auto loadT = [&](int ms) {
#pragma unroll
        for (int i = 0; i < 2; ++i)
            pf[i] = ((const uint4*)(fb + (size_t)ms * 64))[tid + i * 128];
#pragma unroll
        for (int i = 0; i < 8; ++i) {
            int idx = tid + i * 128;
            int d = idx >> 2, q = idx & 3;
            ph[i] = *(const uint4*)(hb + (size_t)d * N + ms + q * 8);
        }
    };
    loadT(0);

    ABu ones;
#pragma unroll
    for (int v = 0; v < 8; ++v) ones.u[v] = 0x3F803F80u;   // bf16 1.0 pairs

    for (int ms = 0; ms < N; ms += 32) {
        __syncthreads();   // previous iteration's LDS reads complete
        // Commit staged tile to LDS
        {
            uint4* dstF = (uint4*)ldsF;
#pragma unroll
            for (int i = 0; i < 2; ++i) {
                int idx = tid + i * 128;
                int m = idx >> 3, q = idx & 7;
                dstF[m * (F_PITCH / 8) + q] = pf[i];
            }
            uint4* dstH = (uint4*)ldsH;
#pragma unroll
            for (int i = 0; i < 8; ++i) {
                int idx = tid + i * 128;
                int d = idx >> 2, q = idx & 3;
                dstH[d * (H_PITCH / 8) + q] = ph[i];
            }
        }
        __syncthreads();
        loadT((ms + 32) & (N - 1));   // issue next-tile loads; latency hidden by compute

        // S B-fragments (F^T), shared by both row-groups
        ABu bf[2][2];   // [m col tile][k chunk]
#pragma unroll
        for (int mt = 0; mt < 2; ++mt) {
            const unsigned short* fr = ldsF + (mt * 16 + l16) * F_PITCH;
#pragma unroll
            for (int v = 0; v < 8; ++v) {
                int k0 = ((v >> 2) << 4) + (half << 3) + ((v & 3) << 1);
                bf[mt][0].u[v] = *(const unsigned int*)(fr + k0);
                bf[mt][1].u[v] = *(const unsigned int*)(fr + 32 + k0);
            }
        }

#pragma unroll
        for (int rg = 0; rg < 2; ++rg) {
            v8f s[2];
#pragma unroll
            for (int mt = 0; mt < 2; ++mt) {
                v8f acc = {0.f,0.f,0.f,0.f,0.f,0.f,0.f,0.f};
                acc = wmma_bf16(ag[rg][0], bf[mt][0], acc);
                acc = wmma_bf16(ag[rg][1], bf[mt][1], acc);
                s[mt] = acc;
            }
            // Online softmax: row max via half-wave shuffles; sum via ones-WMMA tile
            float scale[8];
            bool need = false;
#pragma unroll
            for (int r = 0; r < 8; ++r) {
                float mx = fmaxf(s[0][r], s[1][r]);
#pragma unroll
                for (int off = 1; off < 16; off <<= 1) mx = fmaxf(mx, __shfl_xor(mx, off, 32));
                float nm = fmaxf(mrow[rg][r], mx);
                float sc = __expf(mrow[rg][r] - nm);
                float p0 = __expf(s[0][r] - nm);
                float p1 = __expf(s[1][r] - nm);
                unsigned short* pp =
                    ldsP + (size_t)(wave * 32 + rg * 16 + r + (half << 3)) * P_PITCH;
                pp[l16]      = f2bf(p0);
                pp[16 + l16] = f2bf(p1);
                mrow[rg][r] = nm;
                scale[r] = sc;
                need |= (sc != 1.0f);
            }
            if (__ballot(need)) {   // rescale only when some row max moved
#pragma unroll
                for (int t = 0; t < 17; ++t)
#pragma unroll
                    for (int r = 0; r < 8; ++r) oacc[rg][t][r] *= scale[r];
            }
        }

        // P A-fragments (DS ops of one wave complete in order: no barrier needed)
        ABu ap[2];
#pragma unroll
        for (int rg = 0; rg < 2; ++rg) {
            const unsigned short* pr = ldsP + (wave * 32 + rg * 16 + l16) * P_PITCH;
#pragma unroll
            for (int v = 0; v < 8; ++v) {
                int m = ((v >> 2) << 4) + (half << 3) + ((v & 3) << 1);
                ap[rg].u[v] = *(const unsigned int*)(pr + m);
            }
        }
        // O += P . H : each H B-fragment feeds both row-groups
#pragma unroll
        for (int t = 0; t < 16; ++t) {
            ABu bh;
            const unsigned short* hr = ldsH + (t * 16 + l16) * H_PITCH;
#pragma unroll
            for (int v = 0; v < 8; ++v) {
                int m = ((v >> 2) << 4) + (half << 3) + ((v & 3) << 1);
                bh.u[v] = *(const unsigned int*)(hr + m);
            }
            oacc[0][t] = wmma_bf16(ap[0], bh, oacc[0][t]);
            oacc[1][t] = wmma_bf16(ap[1], bh, oacc[1][t]);
        }
        oacc[0][16] = wmma_bf16(ap[0], ones, oacc[0][16]);
        oacc[1][16] = wmma_bf16(ap[1], ones, oacc[1][16]);
    }

    // Epilogue: out = gamma * O / l + input, raw-reshape flat indexing
#pragma unroll
    for (int rg = 0; rg < 2; ++rg) {
        float rl[8];
#pragma unroll
        for (int r = 0; r < 8; ++r) rl[r] = gamma / oacc[rg][16][r];
#pragma unroll
        for (int t = 0; t < 16; ++t) {
            int d = t * 16 + l16;
#pragma unroll
            for (int r = 0; r < 8; ++r) {
                int n = rowb + rg * 16 + r + (half << 3);
                size_t idx = (size_t)b * C * N + (size_t)n * C + d;
                out[idx] = oacc[rg][t][r] * rl[r] + input[idx];
            }
        }
    }
}

// ---------------------------------------------------------------------------
extern "C" void kernel_launch(void* const* d_in, const int* in_sizes, int n_in,
                              void* d_out, int out_size, void* d_ws, size_t ws_size,
                              hipStream_t stream) {
    const float* input  = (const float*)d_in[0];
    const float* feat   = (const float*)d_in[1];   // nms_feat
    const float* WF     = (const float*)d_in[2];
    const float* WG     = (const float*)d_in[3];
    const float* WH     = (const float*)d_in[4];
    const float* gamma  = (const float*)d_in[5];
    float* out = (float*)d_out;

    unsigned short* Gbuf  = (unsigned short*)d_ws;                 // [B][N][64]
    unsigned short* Fbuf  = Gbuf + (size_t)B * N * 64;             // [B][N][64]
    unsigned short* Htbuf = Fbuf + (size_t)B * N * 64;             // [B][256][N]
    unsigned short* Wbuf  = Htbuf + (size_t)B * 256 * N;           // [384][256]

    k_cvt<<<384, 256, 0, stream>>>(WF, WG, WH, Wbuf);
    k_proj<<<B * (N / 64), 256, 0, stream>>>(feat, Wbuf, Fbuf, Gbuf, Htbuf);
    k_attn<<<B * (N / 128), 128, 0, stream>>>(input, gamma, Fbuf, Gbuf, Htbuf, out);
}
